// AgentModule_36000415875116
// MI455X (gfx1250) — matrix operand order repaired and checked by reading, not compiled, not run on hardware
//
#include <hip/hip_runtime.h>
#include <cstdint>
#include <cstddef>

// ---------------------------------------------------------------------------
// Types for CDNA5 WMMA (wave32): D(f32 16x16) = A(bf16 16x32) x B(bf16 32x16) + C
// ---------------------------------------------------------------------------
typedef __attribute__((ext_vector_type(16))) __bf16 v16bf;
typedef __attribute__((ext_vector_type(8)))  float  v8f;

#define HD   256   // hidden size H
#define NG   768   // 3*H gate width
#define NA   128   // num agents
#define NE   256   // num entities
#define VOC  32    // vocab

__device__ __forceinline__ v8f wmma_bf(v16bf a, v16bf b, v8f c) {
  // v_wmma_f32_16x16x32_bf16
  return __builtin_amdgcn_wmma_f32_16x16x32_bf16(false, a, false, b, (short)0, c,
                                                 false, false);
}

// A fragment (16x32 bf16) from an LDS tile [16][stride].
// Lane L: m = L&15, half = L>>4; holds 16 contiguous K = {k0+half*16 .. +15}.
__device__ __forceinline__ v16bf load_a_lds(const __bf16* sA, int stride, int k0, int lane) {
  int m = lane & 15, hf = lane >> 4;
  const __bf16* p = sA + m * stride;
  int b1 = k0 + hf * 16;
  int b2 = k0 + hf * 16 + 8;
  v16bf a;
#pragma unroll
  for (int i = 0; i < 8; ++i) a[i] = p[b1 + i];
#pragma unroll
  for (int i = 0; i < 8; ++i) a[8 + i] = p[b2 + i];
  return a;
}

// B fragment (32x16 bf16) from row-major W[N,K]; B[k,n] = W[n,k] (NT gemm).
// Lane L: n = n0 + (L&15); holds 16 contiguous K starting at k0 + (L>>4)*16.
__device__ __forceinline__ v16bf load_b_glb(const __bf16* __restrict__ W, int K,
                                            int n0, int k0, int lane) {
  int n = n0 + (lane & 15), kh = lane >> 4;
  return *reinterpret_cast<const v16bf*>(W + (size_t)n * K + k0 + kh * 16);
}

__device__ __forceinline__ float sigmoidf_(float x) { return 1.0f / (1.0f + __expf(-x)); }
__device__ __forceinline__ float eluf_(float x)     { return x > 0.f ? x : expm1f(x); }

// ---------------------------------------------------------------------------
// Fused GRU step, weight-stationary persistent blocks.
// Block: 128 threads (4 waves). blockIdx.y = column group q (0..3): owns gate
// columns {g*256 + q*64 .. +63} for g = r,z,n (triplet-preserving so the gate
// math can be fused). Wave w holds B tiles (u = 4q+w, g = 0..2) for ALL 8
// k-steps in registers (Breg[3][8] = 192 VGPRs) and reuses them across `iters`
// consecutive 16-row blocks.
// Hidden-state bf16 is READ from h_bf and WRITTEN to ho_bf (ping-pong buffers:
// required because the 4 column-group blocks each read all 256 columns of a
// row but write only their 64-column slice). The fp32 master h_f/ho_f is
// updated in place (each block touches only its own row/col slice of it).
// A-tile staging is software-pipelined: next iteration's tile is prefetched
// into registers while the current K-loop runs; the A fragment is double-
// buffered so ds_load latency hides under the WMMAs.
// mode 1: physical-branch x recomputed inline (K_in=5);
// mode 2: gi read from buffer at (row & gimask), input bias already folded.
// ---------------------------------------------------------------------------
__global__ void __launch_bounds__(128) gru_wmma_kernel(
    const __bf16* __restrict__ h_bf, const float* h_f,
    const __bf16* __restrict__ Whh, const float* __restrict__ bhh,
    int mode,
    const float* __restrict__ Wih, const float* __restrict__ bih,
    const float* __restrict__ obs, const float* __restrict__ phy,
    const float* __restrict__ gibuf, int gimask,
    float* ho_f, __bf16* __restrict__ ho_bf, int iters)
{
  __shared__ __bf16 sA[16][HD + 8];
  __shared__ float  sGH[16][3 * 64 + 4];
  const int tid  = threadIdx.x;
  const int wave = tid >> 5, lane = tid & 31;
  const int q    = blockIdx.y;                       // column group 0..3

  // Load resident weight panel: 3 gate tiles x 8 k-steps, once per block.
  v16bf Breg[3][8];
#pragma unroll
  for (int g = 0; g < 3; ++g) {
    int n0 = g * HD + q * 64 + wave * 16;
#pragma unroll
    for (int ks = 0; ks < 8; ++ks)
      Breg[g][ks] = load_b_glb(Whh, HD, n0, ks * 32, lane);
  }

  // Prefetch first A tile (64B per thread) into registers.
  const int sr = tid >> 3, sc = (tid & 7) * 32;
  v16bf t0, t1;
  {
    const __bf16* src = h_bf + (size_t)(blockIdx.x * iters * 16 + sr) * HD + sc;
    t0 = *reinterpret_cast<const v16bf*>(src);
    t1 = *reinterpret_cast<const v16bf*>(src + 16);
  }

  for (int it = 0; it < iters; ++it) {
    const int row0 = (blockIdx.x * iters + it) * 16;

    // Store prefetched tile to LDS.
    {
      __bf16* dst = &sA[sr][sc];
#pragma unroll
      for (int i = 0; i < 16; ++i) dst[i] = t0[i];
#pragma unroll
      for (int i = 0; i < 16; ++i) dst[16 + i] = t1[i];
    }
    __syncthreads();

    // Prefetch next iteration's tile (overlaps K-loop + epilogue; h_bf is
    // read-only this step, so no hazard).
    if (it + 1 < iters) {
      const __bf16* src = h_bf + (size_t)(row0 + 16 + sr) * HD + sc;
      t0 = *reinterpret_cast<const v16bf*>(src);
      t1 = *reinterpret_cast<const v16bf*>(src + 16);
    }

    v8f acc[3];
#pragma unroll
    for (int g = 0; g < 3; ++g)
#pragma unroll
      for (int i = 0; i < 8; ++i) acc[g][i] = 0.0f;

    // K = 256, 32 per WMMA; double-buffered A fragment.
    v16bf a = load_a_lds(&sA[0][0], HD + 8, 0, lane);
#pragma unroll
    for (int ks = 0; ks < 8; ++ks) {
      v16bf an = a;
      if (ks < 7) an = load_a_lds(&sA[0][0], HD + 8, (ks + 1) * 32, lane);
      acc[0] = wmma_bf(a, Breg[0][ks], acc[0]);
      acc[1] = wmma_bf(a, Breg[1][ks], acc[1]);
      acc[2] = wmma_bf(a, Breg[2][ks], acc[2]);
      a = an;
    }

    // Spill gh (fp32) to LDS: local col = g*64 + wave*16 + (lane&15).
    {
      int mrow = (lane >> 4) * 8;
      int cb   = wave * 16 + (lane & 15);
#pragma unroll
      for (int g = 0; g < 3; ++g)
#pragma unroll
        for (int r = 0; r < 8; ++r)
          sGH[mrow + r][g * 64 + cb] = acc[g][r];
    }
    __syncthreads();

    // Epilogue: GRU gate math, 16 rows x 64 hidden cols (j = q*64 + jl).
    for (int idx = tid; idx < 16 * 64; idx += 128) {
      int m = idx >> 6, jl = idx & 63;
      int j = q * 64 + jl;
      int row = row0 + m;
      float ghr = sGH[m][jl]        + bhh[j];
      float ghz = sGH[m][64 + jl]   + bhh[HD + j];
      float ghn = sGH[m][128 + jl]  + bhh[2 * HD + j];
      float gir, giz, gin;
      if (mode == 1) {  // physical branch: x = [obs(a,e,2), physical(e,3)]
        const float* op = obs + (size_t)row * 2;
        float x0 = op[0], x1 = op[1];
        const float* pp = phy + (size_t)(row & (NE - 1)) * 3;
        float x2 = pp[0], x3 = pp[1], x4 = pp[2];
        const float* w0 = Wih + (size_t)j * 5;
        const float* w1 = Wih + (size_t)(HD + j) * 5;
        const float* w2 = Wih + (size_t)(2 * HD + j) * 5;
        gir = bih[j]          + x0*w0[0] + x1*w0[1] + x2*w0[2] + x3*w0[3] + x4*w0[4];
        giz = bih[HD + j]     + x0*w1[0] + x1*w1[1] + x2*w1[2] + x3*w1[3] + x4*w1[4];
        gin = bih[2 * HD + j] + x0*w2[0] + x1*w2[1] + x2*w2[2] + x3*w2[3] + x4*w2[4];
      } else {          // precomputed gi (bias folded); utterance: row & 127
        const float* gp = gibuf + (size_t)(row & gimask) * NG;
        gir = gp[j]; giz = gp[HD + j]; gin = gp[2 * HD + j];
      }
      float r  = sigmoidf_(gir + ghr);
      float z  = sigmoidf_(giz + ghz);
      float n  = tanhf(gin + r * ghn);
      float hp = h_f[(size_t)row * HD + j];
      float h  = (1.f - z) * n + z * hp;
      ho_f[(size_t)row * HD + j]  = h;
      ho_bf[(size_t)row * HD + j] = (__bf16)h;
    }
    // Next iteration's spill happens after its post-stage barrier, so the
    // epilogue reads above are protected; no extra barrier needed here.
  }
}

// ---------------------------------------------------------------------------
// FC layer via WMMA, weight-stationary persistent blocks.
// Block: 128 threads (4 waves). blockIdx.y = column half (0/1): 8 of 16
// n-tiles; wave w holds tiles {h*8 + 2w, h*8 + 2w + 1} for all k-steps in
// registers (Breg[2][8] = 128 VGPRs). out_bf16 = act(A_bf16 @ W_bf16^T + b).
// Same staging pipeline / fragment double-buffering as the GRU kernel.
// ---------------------------------------------------------------------------
__global__ void __launch_bounds__(128) fc_wmma_kernel(
    const __bf16* __restrict__ A, const __bf16* __restrict__ W,
    const float* __restrict__ bias, __bf16* __restrict__ out, int act, int iters)
{
  __shared__ __bf16 sA[16][HD + 8];
  const int tid  = threadIdx.x;
  const int wave = tid >> 5, lane = tid & 31;
  const int h    = blockIdx.y;                       // column half 0..1

  v16bf Breg[2][8];
#pragma unroll
  for (int t = 0; t < 2; ++t) {
    int n0 = (h * 8 + wave * 2 + t) * 16;
#pragma unroll
    for (int ks = 0; ks < 8; ++ks)
      Breg[t][ks] = load_b_glb(W, HD, n0, ks * 32, lane);
  }

  const int sr = tid >> 3, sc = (tid & 7) * 32;
  v16bf t0, t1;
  {
    const __bf16* src = A + (size_t)(blockIdx.x * iters * 16 + sr) * HD + sc;
    t0 = *reinterpret_cast<const v16bf*>(src);
    t1 = *reinterpret_cast<const v16bf*>(src + 16);
  }

  for (int it = 0; it < iters; ++it) {
    const int row0 = (blockIdx.x * iters + it) * 16;
    {
      __bf16* dst = &sA[sr][sc];
#pragma unroll
      for (int i = 0; i < 16; ++i) dst[i] = t0[i];
#pragma unroll
      for (int i = 0; i < 16; ++i) dst[16 + i] = t1[i];
    }
    __syncthreads();
    if (it + 1 < iters) {
      const __bf16* src = A + (size_t)(row0 + 16 + sr) * HD + sc;
      t0 = *reinterpret_cast<const v16bf*>(src);
      t1 = *reinterpret_cast<const v16bf*>(src + 16);
    }

    v8f acc[2];
#pragma unroll
    for (int t = 0; t < 2; ++t)
#pragma unroll
      for (int i = 0; i < 8; ++i) acc[t][i] = 0.0f;

    v16bf a = load_a_lds(&sA[0][0], HD + 8, 0, lane);
#pragma unroll
    for (int ks = 0; ks < 8; ++ks) {
      v16bf an = a;
      if (ks < 7) an = load_a_lds(&sA[0][0], HD + 8, (ks + 1) * 32, lane);
      acc[0] = wmma_bf(a, Breg[0][ks], acc[0]);
      acc[1] = wmma_bf(a, Breg[1][ks], acc[1]);
      a = an;
    }

#pragma unroll
    for (int t = 0; t < 2; ++t) {
      int col  = (h * 8 + wave * 2 + t) * 16 + (lane & 15);
      float bv = bias[col];
#pragma unroll
      for (int r = 0; r < 8; ++r) {
        int row = row0 + (lane >> 4) * 8 + r;
        float v = acc[t][r] + bv;
        if (act) v = eluf_(v);
        out[(size_t)row * HD + col] = (__bf16)v;
      }
    }
    __syncthreads();   // all waves done reading sA before it is re-staged
  }
}

// ---------------------------------------------------------------------------
// goal_pred head: sum((gph @ gp_W2^T + gp_b2)^2) over all 16384 rows, 5 cols.
// ---------------------------------------------------------------------------
__global__ void __launch_bounds__(256) gp2_cost_kernel(
    const __bf16* __restrict__ gph, const float* __restrict__ W2,
    const float* __restrict__ b2, float* cost)
{
  int row = blockIdx.x * 256 + threadIdx.x;
  const __bf16* ar = gph + (size_t)row * HD;
  float local = 0.f;
#pragma unroll 1
  for (int c = 0; c < 5; ++c) {
    const float* w = W2 + c * HD;
    float s = b2[c];
    for (int k = 0; k < HD; ++k) s += (float)ar[k] * w[k];
    local += s * s;
  }
  __shared__ float red[256];
  red[threadIdx.x] = local;
  __syncthreads();
  for (int s = 128; s > 0; s >>= 1) {
    if (threadIdx.x < s) red[threadIdx.x] += red[threadIdx.x + s];
    __syncthreads();
  }
  if (threadIdx.x == 0) atomicAdd(cost, red[0]);
}

// Mean over axis 1 of [128, cnt, 256] bf16 -> x_a[:, dstOff : dstOff+256] fp32.
__global__ void mean_kernel(const __bf16* __restrict__ src, float* __restrict__ dst,
                            int cnt, int dstOff)
{
  int idx = blockIdx.x * blockDim.x + threadIdx.x;
  if (idx >= NA * HD) return;
  int i = idx >> 8, h = idx & (HD - 1);
  const __bf16* p = src + ((size_t)i * cnt) * HD + h;
  float s = 0.f;
  for (int c = 0; c < cnt; ++c) s += (float)p[(size_t)c * HD];
  dst[(size_t)i * NG + dstOff + h] = s / (float)cnt;
}

// Goal-processor GRU (128 rows, discarded mem): g2 = GRU(observed_goals, mem_a).
__global__ void goal_gru_kernel(const float* __restrict__ goals,
    const float* __restrict__ Wih, const float* __restrict__ Whh,
    const float* __restrict__ bih, const float* __restrict__ bhh,
    const float* __restrict__ hA, float* __restrict__ g2)
{
  int idx = blockIdx.x * blockDim.x + threadIdx.x;
  if (idx >= NA * HD) return;
  int rr = idx >> 8, j = idx & (HD - 1);
  const float* x = goals + (size_t)rr * 5;
  const float* h = hA + (size_t)rr * HD;
  float gi[3], gh[3];
#pragma unroll
  for (int g = 0; g < 3; ++g) {
    int c = g * HD + j;
    const float* wi = Wih + (size_t)c * 5;
    float s = bih[c];
#pragma unroll
    for (int k = 0; k < 5; ++k) s += x[k] * wi[k];
    gi[g] = s;
    const float* wh = Whh + (size_t)c * HD;
    float t = bhh[c];
    for (int k = 0; k < HD; ++k) t += h[k] * wh[k];
    gh[g] = t;
  }
  float r = sigmoidf_(gi[0] + gh[0]);
  float z = sigmoidf_(gi[1] + gh[1]);
  float n = tanhf(gi[2] + r * gh[2]);
  g2[idx] = (1.f - z) * n + z * h[j];
}

// Generic small FC (128 rows, fp32): out[r, outOff+c] = act(A[r,:K] . W[c,:K] + b[c])
__global__ void small_fc_kernel(const float* __restrict__ A, const float* __restrict__ W,
    const float* __restrict__ b, float* __restrict__ out,
    int K, int N, int act, int outStride, int outOff)
{
  int idx = blockIdx.x * blockDim.x + threadIdx.x;
  if (idx >= NA * N) return;
  int r = idx / N, c = idx - r * N;
  const float* ar = A + (size_t)r * K;
  const float* wr = W + (size_t)c * K;
  float s = b[c];
  for (int k = 0; k < K; ++k) s += ar[k] * wr[k];
  if (act) s = eluf_(s);
  out[(size_t)r * outStride + outOff + c] = s;
}

// Movement + utterance heads, gumbel softmax, cost accumulation. One block/128 thr.
__global__ void __launch_bounds__(128) heads_kernel(
    const float* __restrict__ mvh, const float* __restrict__ uch,
    const float* __restrict__ mW2, const float* __restrict__ mb2,
    const float* __restrict__ ucW2, const float* __restrict__ ucb2,
    const float* __restrict__ gu, float* cost)
{
  int r = threadIdx.x;
  float local = 0.f;
  const float* mh = mvh + (size_t)r * HD;
#pragma unroll
  for (int c = 0; c < 2; ++c) {
    const float* w = mW2 + c * HD;
    float s = mb2[c];
    for (int k = 0; k < HD; ++k) s += mh[k] * w[k];
    float f = tanhf(s) * 0.2f - 0.1f;   // mv*2*STEP - STEP
    local += f * f;
  }
  const float* uh = uch + (size_t)r * HD;
  float y[VOC];
  float mx = -3.4e38f;
  for (int c = 0; c < VOC; ++c) {
    const float* w = ucW2 + c * HD;
    float s = ucb2[c];
    for (int k = 0; k < HD; ++k) s += uh[k] * w[k];
    float g  = gu[(size_t)r * VOC + c];
    float yv = s - logf(-logf(g + 1e-20f) + 1e-20f);
    y[c] = yv;
    mx = fmaxf(mx, yv);
  }
  float ssum = 0.f;
  for (int c = 0; c < VOC; ++c) { float e = __expf(y[c] - mx); y[c] = e; ssum += e; }
  float inv = 1.f / ssum;
  for (int c = 0; c < VOC; ++c) { float u = y[c] * inv; local += u * u; }
  __shared__ float red[128];
  red[r] = local;
  __syncthreads();
  for (int s = 64; s > 0; s >>= 1) {
    if (r < s) red[r] += red[r + s];
    __syncthreads();
  }
  if (r == 0) atomicAdd(cost, red[0]);
}

__global__ void cvt_bf_kernel(const float* __restrict__ src, __bf16* __restrict__ dst, int n)
{
  int i = blockIdx.x * blockDim.x + threadIdx.x;
  if (i < n) dst[i] = (__bf16)src[i];
}

// ---------------------------------------------------------------------------
extern "C" void kernel_launch(void* const* d_in, const int* in_sizes, int n_in,
                              void* d_out, int out_size, void* d_ws, size_t ws_size,
                              hipStream_t stream) {
  (void)in_sizes; (void)n_in; (void)out_size; (void)ws_size;
  const float* utter  = (const float*)d_in[0];
  const float* obs    = (const float*)d_in[1];
  const float* phys   = (const float*)d_in[2];
  const float* goals  = (const float*)d_in[3];
  const float* gumbel = (const float*)d_in[7];
  const float* u_Wih = (const float*)d_in[8],  *u_Whh = (const float*)d_in[9];
  const float* u_bih = (const float*)d_in[10], *u_bhh = (const float*)d_in[11];
  const float* u_Wfc = (const float*)d_in[12], *u_bfc = (const float*)d_in[13];
  const float* gp_W1 = (const float*)d_in[14], *gp_b1 = (const float*)d_in[15];
  const float* gp_W2 = (const float*)d_in[16], *gp_b2 = (const float*)d_in[17];
  const float* p_Wih = (const float*)d_in[18], *p_Whh = (const float*)d_in[19];
  const float* p_bih = (const float*)d_in[20], *p_bhh = (const float*)d_in[21];
  const float* p_Wfc = (const float*)d_in[22], *p_bfc = (const float*)d_in[23];
  const float* g_Wih = (const float*)d_in[24], *g_Whh = (const float*)d_in[25];
  const float* g_bih = (const float*)d_in[26], *g_bhh = (const float*)d_in[27];
  const float* g_Wfc = (const float*)d_in[28], *g_bfc = (const float*)d_in[29];
  const float* a_Wih = (const float*)d_in[30], *a_Whh = (const float*)d_in[31];
  const float* a_bih = (const float*)d_in[32], *a_bhh = (const float*)d_in[33];
  const float* a_Wfc = (const float*)d_in[34], *a_bfc = (const float*)d_in[35];
  const float* m_W1  = (const float*)d_in[36], *m_b1  = (const float*)d_in[37];
  const float* m_W2  = (const float*)d_in[38], *m_b2  = (const float*)d_in[39];
  const float* uc_W1 = (const float*)d_in[40], *uc_b1 = (const float*)d_in[41];
  const float* uc_W2 = (const float*)d_in[42], *uc_b2 = (const float*)d_in[43];
  float* cost = (float*)d_out;

  // Workspace carve (bump allocator, 256B aligned)
  char* wp = (char*)d_ws;
  auto carve = [&](size_t bytes) -> char* {
    char* r = wp; wp += (bytes + 255) & ~(size_t)255; return r;
  };
  const size_t RU = (size_t)NA * NA;  // 16384 utterance rows
  const size_t RP = (size_t)NA * NE;  // 32768 physical rows
  float*  memU     = (float*) carve(RU * HD * 4);
  float*  memP     = (float*) carve(RP * HD * 4);
  float*  memA     = (float*) carve((size_t)NA * HD * 4);
  __bf16* memU_bfA = (__bf16*)carve(RU * HD * 2);
  __bf16* memU_bfB = (__bf16*)carve(RU * HD * 2);
  __bf16* memP_bfA = (__bf16*)carve(RP * HD * 2);
  __bf16* memP_bfB = (__bf16*)carve(RP * HD * 2);
  __bf16* memA_bfA = (__bf16*)carve((size_t)NA * HD * 2);
  __bf16* memA_bfB = (__bf16*)carve((size_t)NA * HD * 2);
  __bf16* uproc    = (__bf16*)carve(RU * HD * 2);
  __bf16* gph      = (__bf16*)carve(RU * HD * 2);
  __bf16* pproc    = (__bf16*)carve(RP * HD * 2);
  float*  gi_u     = (float*) carve((size_t)NA * NG * 4);
  float*  x_a      = (float*) carve((size_t)NA * NG * 4);
  float*  gi_a     = (float*) carve((size_t)NA * NG * 4);
  float*  g2b      = (float*) carve((size_t)NA * HD * 4);
  float*  procb    = (float*) carve((size_t)NA * HD * 4);
  float*  mvhb     = (float*) carve((size_t)NA * HD * 4);
  float*  uchb     = (float*) carve((size_t)NA * HD * 4);
  __bf16* uWhh_bf  = (__bf16*)carve((size_t)NG * HD * 2);
  __bf16* uWfc_bf  = (__bf16*)carve((size_t)HD * HD * 2);
  __bf16* gpW1_bf  = (__bf16*)carve((size_t)HD * HD * 2);
  __bf16* pWhh_bf  = (__bf16*)carve((size_t)NG * HD * 2);
  __bf16* pWfc_bf  = (__bf16*)carve((size_t)HD * HD * 2);
  __bf16* aWhh_bf  = (__bf16*)carve((size_t)NG * HD * 2);

  // Init: copy mem states (don't mutate inputs), zero cost, convert to bf16.
  hipMemcpyAsync(memU, d_in[4], RU * HD * 4, hipMemcpyDeviceToDevice, stream);
  hipMemcpyAsync(memP, d_in[5], RP * HD * 4, hipMemcpyDeviceToDevice, stream);
  hipMemcpyAsync(memA, d_in[6], (size_t)NA * HD * 4, hipMemcpyDeviceToDevice, stream);
  hipMemsetAsync(d_out, 0, sizeof(float), stream);

  auto cvt = [&](const float* s, __bf16* d, int n) {
    cvt_bf_kernel<<<(n + 255) / 256, 256, 0, stream>>>(s, d, n);
  };
  cvt(memU, memU_bfA, (int)(RU * HD));
  cvt(memP, memP_bfA, (int)(RP * HD));
  cvt(memA, memA_bfA, NA * HD);
  cvt(u_Whh, uWhh_bf, NG * HD);
  cvt(u_Wfc, uWfc_bf, HD * HD);
  cvt(gp_W1, gpW1_bf, HD * HD);
  cvt(p_Whh, pWhh_bf, NG * HD);
  cvt(p_Wfc, pWfc_bf, HD * HD);
  cvt(a_Whh, aWhh_bf, NG * HD);

  // Time-invariant utterance gi: gi_u[j,:] = utterances[j] @ u_Wih^T + u_bih
  small_fc_kernel<<<(NA * NG + 255) / 256, 256, 0, stream>>>(
      utter, u_Wih, u_bih, gi_u, VOC, NG, 0, NG, 0);

  __bf16 *uCur = memU_bfA, *uNxt = memU_bfB;
  __bf16 *pCur = memP_bfA, *pNxt = memP_bfB;
  __bf16 *aCur = memA_bfA, *aNxt = memA_bfB;

  const int ITER = 8;
  for (int t = 0; t < 8; ++t) {
    // Utterance GRU: read uCur (bf16), write uNxt + fp32 master in place.
    gru_wmma_kernel<<<dim3((unsigned)(RU / 16 / ITER), 4), 128, 0, stream>>>(
        uCur, memU, uWhh_bf, u_bhh, 2, nullptr, nullptr, nullptr, nullptr,
        gi_u, NA - 1, memU, uNxt, ITER);
    // utt_proc = ELU(memU_new @ u_Wfc^T + b)
    fc_wmma_kernel<<<dim3((unsigned)(RU / 16 / ITER), 2), 128, 0, stream>>>(
        uNxt, uWfc_bf, u_bfc, uproc, 1, ITER);
    // goal prediction hidden + cost
    fc_wmma_kernel<<<dim3((unsigned)(RU / 16 / ITER), 2), 128, 0, stream>>>(
        uproc, gpW1_bf, gp_b1, gph, 1, ITER);
    gp2_cost_kernel<<<(int)(RU / 256), 256, 0, stream>>>(gph, gp_W2, gp_b2, cost);

    // Physical GRU, x recomputed inline (K_in = 5)
    gru_wmma_kernel<<<dim3((unsigned)(RP / 16 / ITER), 4), 128, 0, stream>>>(
        pCur, memP, pWhh_bf, p_bhh, 1, p_Wih, p_bih, obs, phys,
        nullptr, 0, memP, pNxt, ITER);
    fc_wmma_kernel<<<dim3((unsigned)(RP / 16 / ITER), 2), 128, 0, stream>>>(
        pNxt, pWfc_bf, p_bfc, pproc, 1, ITER);

    // Pooled features -> x_a[:,0:256]=phys_feat, [:,256:512]=utt_feat
    mean_kernel<<<(NA * HD + 255) / 256, 256, 0, stream>>>(pproc, x_a, NE, 0);
    mean_kernel<<<(NA * HD + 255) / 256, 256, 0, stream>>>(uproc, x_a, NA, HD);

    // Goal processor (reads OLD memA fp32, before action-GRU update)
    goal_gru_kernel<<<(NA * HD + 255) / 256, 256, 0, stream>>>(
        goals, g_Wih, g_Whh, g_bih, g_bhh, memA, g2b);
    small_fc_kernel<<<(NA * HD + 255) / 256, 256, 0, stream>>>(
        g2b, g_Wfc, g_bfc, x_a, HD, HD, 1, NG, 2 * HD);

    // Action GRU: gi_a = x_a @ a_Wih^T + a_bih, then WMMA GRU aCur -> aNxt
    small_fc_kernel<<<(NA * NG + 255) / 256, 256, 0, stream>>>(
        x_a, a_Wih, a_bih, gi_a, NG, NG, 0, NG, 0);
    gru_wmma_kernel<<<dim3(1, 4), 128, 0, stream>>>(
        aCur, memA, aWhh_bf, a_bhh, 2, nullptr, nullptr, nullptr, nullptr,
        gi_a, 0x7FFFFFFF, memA, aNxt, ITER);

    // processed = ELU(memA_new @ a_Wfc^T + b); head hiddens; heads + cost
    small_fc_kernel<<<(NA * HD + 255) / 256, 256, 0, stream>>>(
        memA, a_Wfc, a_bfc, procb, HD, HD, 1, HD, 0);
    small_fc_kernel<<<(NA * HD + 255) / 256, 256, 0, stream>>>(
        procb, m_W1, m_b1, mvhb, HD, HD, 1, HD, 0);
    small_fc_kernel<<<(NA * HD + 255) / 256, 256, 0, stream>>>(
        procb, uc_W1, uc_b1, uchb, HD, HD, 1, HD, 0);
    heads_kernel<<<1, 128, 0, stream>>>(
        mvhb, uchb, m_W2, m_b2, uc_W2, uc_b2, gumbel + (size_t)t * NA * VOC, cost);

    // Ping-pong the bf16 hidden-state buffers for the next time step.
    __bf16* tmp;
    tmp = uCur; uCur = uNxt; uNxt = tmp;
    tmp = pCur; pCur = pNxt; pNxt = tmp;
    tmp = aCur; aCur = aNxt; aNxt = tmp;
  }
}
